// SimpleCnn_10222022165194
// MI455X (gfx1250) — matrix operand order repaired and checked by reading, not compiled, not run on hardware
//
#include <hip/hip_runtime.h>

typedef __attribute__((ext_vector_type(16))) __bf16 v16bf;
typedef __attribute__((ext_vector_type(8)))  __bf16 v8bf;
typedef __attribute__((ext_vector_type(8)))  float  v8f;

// ---------------------------------------------------------------------------
// Workspace layout (bytes):
//   h1   : bf16 [32][112][112][64]  (NHWC)        0 .. 51,380,224
//   h2   : bf16 [32][64][56][56]    (NCHW)        51,380,224 .. 64,225,280
//   wp1  : bf16 packed A-frags conv1              64,225,280 .. 64,229,376
//   wp2  : bf16 packed A-frags conv2              64,229,376 .. 64,303,104
//   part : f32  [196][2][256] FC partials         64,303,104 .. 64,704,512
//   wfcp : bf16 packed B-frags FC [6272][32][16]  64,704,512 .. 71,127,040
// ---------------------------------------------------------------------------

// Pack conv weights (fp32) into WMMA A-fragment order, bf16.
// A 16x32 bf16 packing used here (contiguous k): lane L holds M=L&15; klo=(L>>4)*16;
//   elem e -> K = klo+e
__global__ __launch_bounds__(256) void prep_weights(
    const float* __restrict__ w1, const float* __restrict__ w2,
    __bf16* __restrict__ wp1, __bf16* __restrict__ wp2)
{
    int t = blockIdx.x * 256 + threadIdx.x;
    if (t >= 128 + 2304) return;
    if (t < 128) {
        // conv1: 4 chan-tiles x 32 lanes, single K-block (K=27 pad 32)
        int ct = t >> 5, lane = t & 31;
        int M = lane & 15, klo = (lane >> 4) * 16;
        int oc = (ct << 4) + M;
        __bf16* dst = wp1 + t * 16;
#pragma unroll
        for (int e = 0; e < 16; ++e) {
            int K = klo + e;
            float v = (K < 27) ? w1[oc * 27 + K] : 0.0f;   // K = ci*9+kh*3+kw
            dst[e] = (__bf16)v;
        }
    } else {
        // conv2: 4 chan-tiles x 18 K-blocks x 32 lanes; K = (kh*3+kw)*64 + ci
        int u = t - 128;
        int ct = u / 576, rem = u % 576;
        int kb = rem >> 5, lane = rem & 31;
        int M = lane & 15, klo = (lane >> 4) * 16;
        int oc = (ct << 4) + M;
        __bf16* dst = wp2 + u * 16;
#pragma unroll
        for (int e = 0; e < 16; ++e) {
            int eK = klo + e;
            int K = kb * 32 + eK;
            int khkw = K >> 6;          // kh*3+kw
            int ci   = K & 63;
            dst[e] = (__bf16)w2[oc * 576 + ci * 9 + khkw];
        }
    }
}

// Pack wfc (fp32 [200704][10]) into WMMA B-fragment order, bf16:
// wfcp[kb][lane][e], lane: N=lane&15, khalf=lane>>4; elem e -> K=kb*32+khalf*16+e
__global__ __launch_bounds__(256) void prep_wfc(
    const float* __restrict__ wfc, __bf16* __restrict__ wfcp)
{
    int t = blockIdx.x * 256 + threadIdx.x;      // 784*256 = 200704 exactly
    int kb = t >> 5, lane = t & 31;
    int n = lane & 15, khalf = lane >> 4;
    int jn = (n < 10) ? n : 9;
    __bf16* dst = wfcp + (size_t)t * 16;
    int kbase = kb * 32 + khalf * 16;
#pragma unroll
    for (int e = 0; e < 16; ++e) {
        float f = wfc[(size_t)(kbase + e) * 10 + jn];
        dst[e] = (n < 10) ? (__bf16)f : (__bf16)0.0f;
    }
}

// ---------------------------------------------------------------------------
// conv1 + bias + leaky + maxpool2 : x[NCHW f32] -> h1[NHWC bf16]
// Block: 128 thr (4 waves). Tile: 64ch x 4 pre-pool rows x 16 cols.
// Grid: 32 * (224/4) * (224/16) = 25088
// ---------------------------------------------------------------------------
__global__ __launch_bounds__(128) void conv1_fused(
    const float* __restrict__ x, const __bf16* __restrict__ wp1,
    const float* __restrict__ b1, __bf16* __restrict__ h1)
{
    __shared__ __align__(32) __bf16 xt[3 * 6 * 18];     // input patch
    __shared__ __align__(32) __bf16 P[4 * 16 * 32];     // im2col: [rr][n][K]
    int blk = blockIdx.x;
    int b = blk / 784, rem = blk - b * 784;
    int r0 = (rem / 14) * 4, c0 = (rem % 14) * 16;
    int tid = threadIdx.x;

    for (int p = tid; p < 324; p += 128) {
        int ci = p / 108, q = p - ci * 108;
        int rr = q / 18, cc = q - rr * 18;
        int gr = r0 - 1 + rr, gc = c0 - 1 + cc;
        float v = 0.0f;
        if ((unsigned)gr < 224u && (unsigned)gc < 224u)
            v = x[(((size_t)b * 3 + ci) * 224 + gr) * 224 + gc];
        xt[p] = (__bf16)v;
    }
    __syncthreads();

    // Build im2col patch matrix: 64 threads, one (rr,n) pair each.
    // K is a compile-time constant -> ci/kh/kw fold to immediate LDS offsets.
    if (tid < 64) {
        int rr = tid >> 4, n = tid & 15;
        const __bf16* src = xt + rr * 18 + n;
        v16bf lo16, hi16;
#pragma unroll
        for (int e = 0; e < 16; ++e) {
            constexpr int dummy = 0; (void)dummy;
            const int K = e;                       // compile-time
            const int ci = K / 9, kh = (K % 9) / 3, kw = K % 3;
            lo16[e] = src[ci * 108 + kh * 18 + kw];
        }
#pragma unroll
        for (int e = 0; e < 16; ++e) {
            const int K = e + 16;                  // compile-time
            __bf16 v = (__bf16)0.0f;
            if (K < 27) {
                const int ci = K / 9, kh = (K % 9) / 3, kw = K % 3;
                v = src[ci * 108 + kh * 18 + kw];
            }
            hi16[e] = v;
        }
        *(v16bf*)(P + (tid << 5))      = lo16;
        *(v16bf*)(P + (tid << 5) + 16) = hi16;
    }
    __syncthreads();

    int lane = tid & 31, ct = tid >> 5;
    int n = lane & 15, hi = lane >> 4;
    int kbase = hi << 4;

    v16bf a = *(const v16bf*)(wp1 + (((ct << 5) + lane) << 4));

    v8f acc[4];
#pragma unroll
    for (int rr = 0; rr < 4; ++rr) {
        v16bf bv = *(const v16bf*)(P + (((rr << 4) + n) << 5) + kbase);
        v8f z = {};
        acc[rr] = __builtin_amdgcn_wmma_f32_16x16x32_bf16(
            false, a, false, bv, (short)0, z, false, false);
    }

    // bias + leaky + 2x2 maxpool, store NHWC bf16 (8 consecutive channels)
#pragma unroll
    for (int rp = 0; rp < 2; ++rp) {
        v8bf outv;
#pragma unroll
        for (int v = 0; v < 8; ++v) {
            float bias = b1[(ct << 4) + (hi << 3) + v];
            float y0 = acc[2 * rp][v] + bias;     y0 = fmaxf(y0, 0.01f * y0);
            float y1 = acc[2 * rp + 1][v] + bias; y1 = fmaxf(y1, 0.01f * y1);
            float m  = fmaxf(y0, y1);
            float o  = __shfl_xor(m, 1);
            outv[v]  = (__bf16)fmaxf(m, o);
        }
        if ((lane & 1) == 0) {
            int pr = (r0 >> 1) + rp, pc = (c0 >> 1) + (n >> 1);
            *(v8bf*)(h1 + ((((size_t)b * 112 + pr) * 112 + pc) << 6)
                        + (ct << 4) + (hi << 3)) = outv;
        }
    }
}

// ---------------------------------------------------------------------------
// conv2 + bias + leaky + maxpool2 : h1[NHWC bf16] -> h2[NCHW bf16]
// Block: 128 thr. Tile: 64ch x 4 pre-pool rows x 16 cols. Grid: 32*28*7=6272
// ---------------------------------------------------------------------------
__global__ __launch_bounds__(128) void conv2_fused(
    const __bf16* __restrict__ h1, const __bf16* __restrict__ wp2,
    const float* __restrict__ b2, __bf16* __restrict__ h2)
{
    __shared__ __align__(32) __bf16 xt2[6 * 18 * 64];   // 13,824 B
    int blk = blockIdx.x;
    int b = blk / 196, rem = blk - b * 196;
    int r0 = (rem / 7) * 4, c0 = (rem % 7) * 16;
    int tid = threadIdx.x;

    // stage 6x18x64 bf16 input tile via b128 chunks (8 bf16 each)
    for (int c = tid; c < 864; c += 128) {
        int pos = c >> 3, cg = c & 7;
        int rr = pos / 18, cc = pos - rr * 18;
        int gr = r0 - 1 + rr, gc = c0 - 1 + cc;
        uint4 val = make_uint4(0u, 0u, 0u, 0u);
        if ((unsigned)gr < 112u && (unsigned)gc < 112u)
            val = *(const uint4*)(h1 + ((((size_t)b * 112 + gr) * 112 + gc) << 6)
                                     + (cg << 3));
        *(uint4*)(&xt2[(pos << 6) + (cg << 3)]) = val;
    }
    __syncthreads();

    int lane = tid & 31, ct = tid >> 5;
    int n = lane & 15, hi = lane >> 4;

    v8f acc[4] = {};
    const __bf16* wpw = wp2 + (((size_t)ct * 18) << 9);   // ct*18*32*16
#pragma unroll
    for (int kb = 0; kb < 18; ++kb) {
        v16bf a = *(const v16bf*)(wpw + (((kb << 5) + lane) << 4));
        int khkw = kb >> 1;
        int kh = khkw / 3, kw = khkw - kh * 3;
        int cio = ((kb & 1) << 5) + (hi << 4);
#pragma unroll
        for (int rr = 0; rr < 4; ++rr) {
            v16bf bv = *(const v16bf*)(&xt2[(((rr + kh) * 18 + n + kw) << 6) + cio]);
            acc[rr] = __builtin_amdgcn_wmma_f32_16x16x32_bf16(
                false, a, false, bv, (short)0, acc[rr], false, false);
        }
    }

    // bias + leaky + 2x2 maxpool, store NCHW bf16 (matches reshape order)
#pragma unroll
    for (int rp = 0; rp < 2; ++rp) {
        int pr = (r0 >> 1) + rp, pc = (c0 >> 1) + (n >> 1);
#pragma unroll
        for (int v = 0; v < 8; ++v) {
            int ch = (ct << 4) + (hi << 3) + v;
            float bias = b2[ch];
            float y0 = acc[2 * rp][v] + bias;     y0 = fmaxf(y0, 0.01f * y0);
            float y1 = acc[2 * rp + 1][v] + bias; y1 = fmaxf(y1, 0.01f * y1);
            float m  = fmaxf(y0, y1);
            float o  = __shfl_xor(m, 1);
            if ((lane & 1) == 0)
                h2[(((size_t)b * 64 + ch) * 56 + pr) * 56 + pc] = (__bf16)fmaxf(m, o);
        }
    }
}

// ---------------------------------------------------------------------------
// FC split-K partials: h2[32 x 200704 bf16] @ wfcp (packed bf16 B-frags)
// 392 waves: (chunk 0..195) x (batch tile 0..1); each wave 32 K-blocks of 32.
// ---------------------------------------------------------------------------
__global__ __launch_bounds__(128) void fc_partial(
    const __bf16* __restrict__ h2, const __bf16* __restrict__ wfcp,
    float* __restrict__ part)
{
    int gw = blockIdx.x * 4 + (threadIdx.x >> 5);
    int lane = threadIdx.x & 31;
    int bt = gw & 1, chunk = gw >> 1;

    int m = lane & 15, hi = lane >> 4;
    int klo = hi << 4;
    const __bf16* arow = h2 + (size_t)(bt * 16 + m) * 200704;
    const __bf16* brow = wfcp + (((size_t)chunk * 32) << 9) + ((size_t)lane << 4);

    v8f acc = {};
    for (int i = 0; i < 32; ++i) {
        int kstart = (chunk * 32 + i) * 32;
        v8bf lo  = *(const v8bf*)(arow + kstart + klo);
        v8bf hi8 = *(const v8bf*)(arow + kstart + klo + 8);
        v16bf a;
#pragma unroll
        for (int e = 0; e < 8; ++e) { a[e] = lo[e]; a[e + 8] = hi8[e]; }

        v16bf bv = *(const v16bf*)(brow + ((size_t)i << 9));
        acc = __builtin_amdgcn_wmma_f32_16x16x32_bf16(
            false, a, false, bv, (short)0, acc, false, false);
    }

    float* P = part + ((size_t)(chunk * 2 + bt) << 8);
#pragma unroll
    for (int v = 0; v < 8; ++v)
        P[((v + (hi << 3)) << 4) + m] = acc[v];
}

// Deterministic reduction of FC partials + bias -> out[32][10] f32
__global__ __launch_bounds__(320) void fc_finish(
    const float* __restrict__ part, const float* __restrict__ bfc,
    float* __restrict__ out)
{
    int tid = threadIdx.x;
    if (tid >= 320) return;
    int b = tid / 10, j = tid - b * 10;
    int bt = b >> 4, m = b & 15;
    float s = bfc[j];
    const float* base = part + (size_t)bt * 256 + m * 16 + j;
    for (int c = 0; c < 196; ++c) s += base[(size_t)c * 512];
    out[tid] = s;
}

// ---------------------------------------------------------------------------
extern "C" void kernel_launch(void* const* d_in, const int* in_sizes, int n_in,
                              void* d_out, int out_size, void* d_ws, size_t ws_size,
                              hipStream_t stream) {
    const float* x   = (const float*)d_in[0];
    const float* w1  = (const float*)d_in[1];
    const float* b1  = (const float*)d_in[2];
    const float* w2  = (const float*)d_in[3];
    const float* b2  = (const float*)d_in[4];
    const float* wfc = (const float*)d_in[5];
    const float* bfc = (const float*)d_in[6];
    float* out = (float*)d_out;

    char* ws = (char*)d_ws;
    __bf16* h1   = (__bf16*)(ws);                    // 51,380,224 B
    __bf16* h2   = (__bf16*)(ws + 51380224);         // 12,845,056 B
    __bf16* wp1  = (__bf16*)(ws + 64225280);         //      4,096 B
    __bf16* wp2  = (__bf16*)(ws + 64229376);         //     73,728 B
    float*  part = (float*) (ws + 64303104);         //    401,408 B
    __bf16* wfcp = (__bf16*)(ws + 64704512);         //  6,422,528 B

    prep_weights<<<10, 256, 0, stream>>>(w1, w2, wp1, wp2);
    prep_wfc<<<784, 256, 0, stream>>>(wfc, wfcp);
    conv1_fused<<<25088, 128, 0, stream>>>(x, wp1, b1, h1);
    conv2_fused<<<6272, 128, 0, stream>>>(h1, wp2, b2, h2);
    fc_partial<<<98, 128, 0, stream>>>(h2, wfcp, part);
    fc_finish<<<1, 320, 0, stream>>>(part, bfc, out);
}
